// AttentionBlock_74294344286497
// MI455X (gfx1250) — matrix operand order repaired and checked by reading, not compiled, not run on hardware
//
#include <hip/hip_runtime.h>
#include <hip/hip_bf16.h>

// ---------------------------------------------------------------------------
// CDNA5 WMMA (wave32): D(f32 16x16) = A(bf16 16x32) x B(bf16 32x16) + C
// ---------------------------------------------------------------------------
typedef __bf16 bf16x16 __attribute__((ext_vector_type(16)));
typedef __bf16 bf16x8  __attribute__((ext_vector_type(8)));
typedef float  f32x8   __attribute__((ext_vector_type(8)));

struct alignas(16) Pack16 { unsigned int v[4]; };   // 16-byte staging copies

__device__ __forceinline__ __bf16 f2bf(float f) {
    union { float f; unsigned u; } a; a.f = f;
    unsigned r = (a.u + 0x7fffu + ((a.u >> 16) & 1u)) >> 16;   // RNE
    union { unsigned short s; __bf16 b; } o; o.s = (unsigned short)r;
    return o.b;
}

__device__ __forceinline__ f32x8 wmma_bf16(bf16x16 a, bf16x16 b, f32x8 c) {
    return __builtin_amdgcn_wmma_f32_16x16x32_bf16(false, a, false, b, (short)0, c, false, false);
}

__device__ __forceinline__ bf16x16 frag2(const __bf16* p0, const __bf16* p1) {
    bf16x8 lo = *(const bf16x8*)p0;
    bf16x8 hi = *(const bf16x8*)p1;
    return __builtin_shufflevector(lo, hi, 0,1,2,3,4,5,6,7,8,9,10,11,12,13,14,15);
}

// A-frag from [m][k] row-major tile: f[i<8]=k(8h+i), f[8+j]=k(16+8h+j); m=lane&15.
__device__ __forceinline__ bf16x16 load_a(const __bf16* base, int mstride) {
    const int lane = threadIdx.x & 31;
    const __bf16* p = base + (long)(lane & 15) * mstride + 8 * (lane >> 4);
    return frag2(p, p + 16);
}

// B-frag from K-contiguous layout (element (k,n) at base[n*nstride+k]): f[i<8]=k(8h+i), f[8+j]=k(16+8h+j).
__device__ __forceinline__ bf16x16 load_b(const __bf16* base, int nstride) {
    const int lane = threadIdx.x & 31;
    const __bf16* p = base + (long)(lane & 15) * nstride + 8 * (lane >> 4);
    return frag2(p, p + 16);
}

// ---------------------------------------------------------------------------
#define Bsz 16
#define Cch 512
#define Npos 1024
#define NH 8
#define HD 64
#define GROUPS 8
#define SCALE 0.125f   // 1/sqrt(64), applied to BOTH q and k as in reference

// ---------------------------------------------------------------------------
// 1) GroupNorm statistics
// ---------------------------------------------------------------------------
__global__ __launch_bounds__(256) void k_stats(const float* __restrict__ x,
                                               float* __restrict__ stats) {
    const int bg = blockIdx.x;
    const float* p = x + (long)bg * 65536;
    float s = 0.f, s2 = 0.f;
    for (int i = threadIdx.x; i < 65536; i += 256) { float v = p[i]; s += v; s2 += v * v; }
    __shared__ float sh0[256], sh1[256];
    sh0[threadIdx.x] = s; sh1[threadIdx.x] = s2;
    __syncthreads();
    for (int off = 128; off; off >>= 1) {
        if (threadIdx.x < off) { sh0[threadIdx.x] += sh0[threadIdx.x + off];
                                 sh1[threadIdx.x] += sh1[threadIdx.x + off]; }
        __syncthreads();
    }
    if (threadIdx.x == 0) {
        float mean = sh0[0] * (1.0f / 65536.0f);
        float var  = sh1[0] * (1.0f / 65536.0f) - mean * mean;
        stats[2 * bg]     = mean;
        stats[2 * bg + 1] = rsqrtf(var + 1e-5f);
    }
}

// ---------------------------------------------------------------------------
// 2) Normalize + affine + bf16, transposed output xn_t[b][n][c] via LDS tile
// ---------------------------------------------------------------------------
#define TLD 72   // 64 + 8 pad, rows stay 16B aligned (144B stride)
__global__ __launch_bounds__(256) void k_norm(const float* __restrict__ x,
                                              const float* __restrict__ stats,
                                              const float* __restrict__ nw,
                                              const float* __restrict__ nb,
                                              __bf16* __restrict__ xnt) {
    const int b  = blockIdx.z;
    const int c0 = blockIdx.y * 64;
    const int n0 = blockIdx.x * 64;
    __shared__ __align__(16) __bf16 T[64 * TLD];
    const int tid = threadIdx.x;
    const int nl = tid & 63;
    const int cb = tid >> 6;
#pragma unroll
    for (int p = 0; p < 16; ++p) {
        int cl = p * 4 + cb;
        int c  = c0 + cl;
        int bg = b * GROUPS + (c >> 6);
        float m = stats[2 * bg], r = stats[2 * bg + 1];
        float v = x[((long)b * Cch + c) * Npos + n0 + nl];
        T[nl * TLD + cl] = f2bf((v - m) * r * nw[c] + nb[c]);
    }
    __syncthreads();
    const int nr  = tid >> 2;
    const int sg0 = (tid & 3) * 2;
#pragma unroll
    for (int j = 0; j < 2; ++j) {
        int sg = sg0 + j;
        *(Pack16*)(xnt + ((long)b * Npos + n0 + nr) * Cch + c0 + sg * 8) =
            *(const Pack16*)(T + nr * TLD + sg * 8);
    }
}

// ---------------------------------------------------------------------------
// 3) f32 -> bf16 weight conversion
// ---------------------------------------------------------------------------
__global__ __launch_bounds__(256) void k_cvt(const float* __restrict__ src,
                                             __bf16* __restrict__ dst, int n) {
    int i = blockIdx.x * 256 + threadIdx.x;
    if (i < n) dst[i] = f2bf(src[i]);
}

// ---------------------------------------------------------------------------
// Shared double-buffered 128x64 GEMM mainloop (K = 512 staged 32 at a time).
// A [o][c] k-contiguous rows; B (k-contiguous per column) [n][c].
// Wave w owns m-subtile w (16 rows) x 4 n-subtiles -> 4 accumulators.
// One barrier per K-step; next tile staged into the other LDS buffer while
// WMMAs consume the current one.
// ---------------------------------------------------------------------------
__device__ __forceinline__ void gemm_128x64(const __bf16* __restrict__ Ab,
                                            const __bf16* __restrict__ Bb,
                                            __bf16 (*As)[128 * 32],
                                            __bf16 (*Bs)[64 * 32],
                                            f32x8 acc[4]) {
    const int tid = threadIdx.x, wave = tid >> 5;
    const int ra = tid >> 1, sa = (tid & 1) * 2;   // A: 2 Pack16 / thread
    const int rb = tid >> 2, sb = tid & 3;         // B: 1 Pack16 / thread

    *(Pack16*)&As[0][ra * 32 + sa * 8] = *(const Pack16*)(Ab + (long)ra * Cch + sa * 8);
    *(Pack16*)&As[0][ra * 32 + (sa + 1) * 8] = *(const Pack16*)(Ab + (long)ra * Cch + (sa + 1) * 8);
    *(Pack16*)&Bs[0][rb * 32 + sb * 8] = *(const Pack16*)(Bb + (long)rb * Cch + sb * 8);
    __syncthreads();

    for (int k0 = 0; k0 < Cch; k0 += 32) {
        const int cur = (k0 >> 5) & 1, nxt = cur ^ 1;
        const bool more = (k0 + 32) < Cch;
        Pack16 na0, na1, nb0;
        if (more) {
            na0 = *(const Pack16*)(Ab + (long)ra * Cch + k0 + 32 + sa * 8);
            na1 = *(const Pack16*)(Ab + (long)ra * Cch + k0 + 32 + (sa + 1) * 8);
            nb0 = *(const Pack16*)(Bb + (long)rb * Cch + k0 + 32 + sb * 8);
            if (k0 + 64 < Cch) {
                __builtin_prefetch(Ab + (long)ra * Cch + k0 + 64 + sa * 8, 0, 3);
                __builtin_prefetch(Bb + (long)rb * Cch + k0 + 64 + sb * 8, 0, 3);
            }
        }
        bf16x16 a = load_a(&As[cur][wave * 16 * 32], 32);
#pragma unroll
        for (int t = 0; t < 4; ++t) {
            bf16x16 bb = load_b(&Bs[cur][t * 16 * 32], 32);
            acc[t] = wmma_bf16(a, bb, acc[t]);
        }
        if (more) {
            *(Pack16*)&As[nxt][ra * 32 + sa * 8] = na0;
            *(Pack16*)&As[nxt][ra * 32 + (sa + 1) * 8] = na1;
            *(Pack16*)&Bs[nxt][rb * 32 + sb * 8] = nb0;
        }
        __syncthreads();
    }
}

// ---------------------------------------------------------------------------
// 4) QKV GEMM + scatter to q^T/k^T (scaled) and v
// ---------------------------------------------------------------------------
__global__ __launch_bounds__(256) void k_qkv(const __bf16* __restrict__ W,
                                             const __bf16* __restrict__ xnt,
                                             const float* __restrict__ bias,
                                             __bf16* __restrict__ qt,
                                             __bf16* __restrict__ kt,
                                             __bf16* __restrict__ vs) {
    const int b  = blockIdx.z;
    const int o0 = blockIdx.y * 128;
    const int n0 = blockIdx.x * 64;
    __shared__ __align__(16) __bf16 As[2][128 * 32];
    __shared__ __align__(16) __bf16 Bs[2][64 * 32];
    const int wave = threadIdx.x >> 5, lane = threadIdx.x & 31;
    f32x8 acc[4] = { {}, {}, {}, {} };

    gemm_128x64(W + (long)o0 * Cch,
                xnt + (long)b * Npos * Cch + (long)n0 * Cch, As, Bs, acc);

    const int half = lane >> 4, nn = lane & 15;
#pragma unroll
    for (int t = 0; t < 4; ++t) {
#pragma unroll
        for (int v = 0; v < 8; ++v) {
            int m = v + 8 * half;
            int o = o0 + wave * 16 + m;
            int n = n0 + t * 16 + nn;
            float val = acc[t][v] + bias[o];
            int sec = o >> 9;            // 0=q, 1=k, 2=v
            int oc = o & 511, h = oc >> 6, dd = oc & 63;
            long bh = (long)b * NH + h;
            if (sec == 0)      qt[(bh * Npos + n) * HD + dd] = f2bf(val * SCALE);
            else if (sec == 1) kt[(bh * Npos + n) * HD + dd] = f2bf(val * SCALE);
            else               vs[(bh * HD + dd) * Npos + n] = f2bf(val);
        }
    }
}

// ---------------------------------------------------------------------------
// 5) Fused attention, per (b, h, 16-query-row block).
//    Scores in VGPRs (8 tiles/wave); softmax via shfl_xor + 16x8 LDS reduce;
//    normalized bf16 P (32KB LDS) feeds PV WMMA A-fragments as b128 loads.
// ---------------------------------------------------------------------------
__global__ __launch_bounds__(256) void k_attn(const __bf16* __restrict__ qt,
                                              const __bf16* __restrict__ kt,
                                              const __bf16* __restrict__ vs,
                                              __bf16* __restrict__ aot) {
    const int rb = blockIdx.x;
    const int h  = blockIdx.y;
    const int b  = blockIdx.z;
    const long bh = (long)b * NH + h;
    const __bf16* q  = qt + bh * Npos * HD;
    const __bf16* kk = kt + bh * Npos * HD;
    const __bf16* vv = vs + bh * HD * Npos;
    __shared__ __align__(16) __bf16 P16[16 * Npos];   // 32 KB
    __shared__ float red[16][8];                      // [row][wave]
    const int tid = threadIdx.x, wave = tid >> 5, lane = tid & 31;
    const int half = lane >> 4, nn = lane & 15;
    const int row0 = rb * 16;

    bf16x16 a0 = load_a(q + (long)row0 * HD,      HD);   // d = 0..31
    bf16x16 a1 = load_a(q + (long)row0 * HD + 32, HD);   // d = 32..63

    f32x8 acc[8];
#pragma unroll
    for (int t = 0; t < 8; ++t) {
        int col0 = wave * 128 + t * 16;
        bf16x16 b0 = load_b(kk + (long)col0 * HD,      HD);
        bf16x16 b1 = load_b(kk + (long)col0 * HD + 32, HD);
        f32x8 c = {};
        c = wmma_bf16(a0, b0, c);
        c = wmma_bf16(a1, b1, c);
        acc[t] = c;
    }

    // ---- row max ----
    float rmx[8];
#pragma unroll
    for (int v = 0; v < 8; ++v) {
        float m = acc[0][v];
#pragma unroll
        for (int t = 1; t < 8; ++t) m = fmaxf(m, acc[t][v]);
#pragma unroll
        for (int msk = 1; msk < 16; msk <<= 1) m = fmaxf(m, __shfl_xor(m, msk, 32));
        rmx[v] = m;
    }
    if (nn == 0) {
#pragma unroll
        for (int v = 0; v < 8; ++v) red[v + 8 * half][wave] = rmx[v];
    }
    __syncthreads();
    float rowmax[8];
#pragma unroll
    for (int v = 0; v < 8; ++v) {
        float m = red[v + 8 * half][0];
#pragma unroll
        for (int w = 1; w < 8; ++w) m = fmaxf(m, red[v + 8 * half][w]);
        rowmax[v] = m;
    }
    // ---- exp + row sum ----
    float rsm[8];
#pragma unroll
    for (int v = 0; v < 8; ++v) rsm[v] = 0.f;
#pragma unroll
    for (int t = 0; t < 8; ++t)
#pragma unroll
        for (int v = 0; v < 8; ++v) {
            float e = __expf(acc[t][v] - rowmax[v]);
            acc[t][v] = e; rsm[v] += e;
        }
#pragma unroll
    for (int v = 0; v < 8; ++v)
#pragma unroll
        for (int msk = 1; msk < 16; msk <<= 1) rsm[v] += __shfl_xor(rsm[v], msk, 32);
    __syncthreads();
    if (nn == 0) {
#pragma unroll
        for (int v = 0; v < 8; ++v) red[v + 8 * half][wave] = rsm[v];
    }
    __syncthreads();
    float rinv[8];
#pragma unroll
    for (int v = 0; v < 8; ++v) {
        float sum = 0.f;
#pragma unroll
        for (int w = 0; w < 8; ++w) sum += red[v + 8 * half][w];
        rinv[v] = 1.0f / sum;
    }
    // ---- write normalized P (bf16, [m][1024]) ----
#pragma unroll
    for (int t = 0; t < 8; ++t) {
        int col0 = wave * 128 + t * 16;
#pragma unroll
        for (int v = 0; v < 8; ++v)
            P16[(v + 8 * half) * Npos + col0 + nn] = f2bf(acc[t][v] * rinv[v]);
    }
    __syncthreads();

    // ---- PV: waves 0..3 take 16-wide dd tiles ----
    if (wave < 4) {
        int dd0 = wave * 16;
        f32x8 o = {};
        for (int kc = 0; kc < Npos; kc += 32) {
            bf16x16 a  = load_a(P16 + kc, Npos);
            bf16x16 bb = load_b(vv + (long)dd0 * Npos + kc, Npos);
            o = wmma_bf16(a, bb, o);
        }
#pragma unroll
        for (int v = 0; v < 8; ++v) {
            int m = v + 8 * half;
            int c = h * HD + dd0 + nn;
            aot[((long)b * Npos + row0 + m) * Cch + c] = f2bf(o[v]);
        }
    }
}

// ---------------------------------------------------------------------------
// 6) Projection GEMM + bias + residual (fp32 out, [b][c][n])
// ---------------------------------------------------------------------------
__global__ __launch_bounds__(256) void k_proj(const __bf16* __restrict__ Wp,
                                              const __bf16* __restrict__ aot,
                                              const float* __restrict__ bias,
                                              const float* __restrict__ x,
                                              float* __restrict__ out) {
    const int b  = blockIdx.z;
    const int o0 = blockIdx.y * 128;
    const int n0 = blockIdx.x * 64;
    __shared__ __align__(16) __bf16 As[2][128 * 32];
    __shared__ __align__(16) __bf16 Bs[2][64 * 32];
    const int wave = threadIdx.x >> 5, lane = threadIdx.x & 31;
    f32x8 acc[4] = { {}, {}, {}, {} };

    gemm_128x64(Wp + (long)o0 * Cch,
                aot + (long)b * Npos * Cch + (long)n0 * Cch, As, Bs, acc);

    const int half = lane >> 4, nn = lane & 15;
#pragma unroll
    for (int t = 0; t < 4; ++t) {
#pragma unroll
        for (int v = 0; v < 8; ++v) {
            int m = v + 8 * half;
            int o = o0 + wave * 16 + m;
            int n = n0 + t * 16 + nn;
            long idx = ((long)b * Cch + o) * Npos + n;
            out[idx] = acc[t][v] + bias[o] + x[idx];
        }
    }
}

// ---------------------------------------------------------------------------
// Host-side launch
// ---------------------------------------------------------------------------
extern "C" void kernel_launch(void* const* d_in, const int* in_sizes, int n_in,
                              void* d_out, int out_size, void* d_ws, size_t ws_size,
                              hipStream_t stream) {
    const float* x      = (const float*)d_in[0];
    const float* norm_w = (const float*)d_in[1];
    const float* norm_b = (const float*)d_in[2];
    const float* qkv_w  = (const float*)d_in[3];
    const float* qkv_b  = (const float*)d_in[4];
    const float* proj_w = (const float*)d_in[5];
    const float* proj_b = (const float*)d_in[6];
    float* out = (float*)d_out;

    char* w = (char*)d_ws;
    float*  stats = (float*)w;  w += 4096;
    __bf16* xnt   = (__bf16*)w; w += (size_t)Bsz * Cch * Npos * 2;   // [b][n][c]
    __bf16* qw    = (__bf16*)w; w += (size_t)3 * Cch * Cch * 2;
    __bf16* pw    = (__bf16*)w; w += (size_t)Cch * Cch * 2;
    __bf16* qt    = (__bf16*)w; w += (size_t)Bsz * Cch * Npos * 2;   // [b,h,n,d]
    __bf16* kt    = (__bf16*)w; w += (size_t)Bsz * Cch * Npos * 2;   // [b,h,n,d]
    __bf16* vsc   = (__bf16*)w; w += (size_t)Bsz * Cch * Npos * 2;   // [b,h,d,n]
    __bf16* aot   = (__bf16*)w; w += (size_t)Bsz * Cch * Npos * 2;   // [b][n][c]

    k_stats<<<Bsz * GROUPS, 256, 0, stream>>>(x, stats);
    k_norm<<<dim3(Npos / 64, Cch / 64, Bsz), 256, 0, stream>>>(x, stats, norm_w, norm_b, xnt);
    k_cvt<<<(3 * Cch * Cch) / 256, 256, 0, stream>>>(qkv_w, qw, 3 * Cch * Cch);
    k_cvt<<<(Cch * Cch) / 256, 256, 0, stream>>>(proj_w, pw, Cch * Cch);
    k_qkv<<<dim3(Npos / 64, (3 * Cch) / 128, Bsz), 256, 0, stream>>>(qw, xnt, qkv_b, qt, kt, vsc);
    k_attn<<<dim3(Npos / 16, NH, Bsz), 256, 0, stream>>>(qt, kt, vsc, aot);
    k_proj<<<dim3(Npos / 64, Cch / 128, Bsz), 256, 0, stream>>>(pw, aot, proj_b, x, out);
}
